// FeaturePropagation_28398323761384
// MI455X (gfx1250) — matrix operand order repaired and checked by reading, not compiled, not run on hardware
//
#include <hip/hip_runtime.h>

// ---------------------------------------------------------------------------
// FeaturePropagation (PointNet++ FP layer) for MI455X / gfx1250
// B=16, N1=1024, N2=4096, C1=256, C2=128, Cin=384, H1=H2=256, M=B*N2=65536
// ---------------------------------------------------------------------------

#define B_    16
#define N1_   1024
#define N2_   4096
#define C1_   256
#define C2_   128
#define CIN_  384
#define H_    256
#define M_    65536
#define RED_BLOCKS 256

typedef __bf16 bf16x16 __attribute__((ext_vector_type(16)));
typedef float  f32x8   __attribute__((ext_vector_type(8)));

// round-to-nearest-even f32 -> bf16 (raw u16)
__device__ __forceinline__ unsigned short f2bf(float f) {
    unsigned u = __float_as_uint(f);
    u += 0x7FFFu + ((u >> 16) & 1u);
    return (unsigned short)(u >> 16);
}

// assemble a 16-element bf16 fragment from two 16-byte chunks
__device__ __forceinline__ bf16x16 ld_frag(const unsigned short* p0,
                                           const unsigned short* p1) {
    union { uint4 u[2]; bf16x16 v; } r;
    r.u[0] = *(const uint4*)p0;
    r.u[1] = *(const uint4*)p1;
    return r.v;
}

// ---------------------------------------------------------------------------
// Tensor Data Mover: 2D tile (rows x kElems bf16, row stride = kElems) -> LDS
// D# per CDNA5 ISA ch.8: group0 {count=1, lds_addr, global_addr, type=2},
// group1 {data_size=2B, tensor_dim0/1, tile_dim0/1, tensor_dim0_stride}.
// This toolchain exposes the 6-arg builtin (g0, g1, g2, g3, g_extra, cpol).
// ---------------------------------------------------------------------------
#if __has_builtin(__builtin_amdgcn_tensor_load_to_lds)
#define FP_HAVE_TDM 1
typedef unsigned int u32x4_t __attribute__((ext_vector_type(4)));
typedef int          i32x4_t __attribute__((ext_vector_type(4)));
typedef int          i32x8_t __attribute__((ext_vector_type(8)));

__device__ __forceinline__ void tdm_load_2d(const void* gsrc, unsigned lds_off,
                                            int kElems, int rows) {
    const unsigned long long ga = (unsigned long long)gsrc;
    u32x4_t g0;
    g0[0] = 1u;                                              // count=1 (valid user D#)
    g0[1] = lds_off;                                         // lds_addr (bytes)
    g0[2] = (unsigned)(ga & 0xFFFFFFFFu);                    // global_addr[31:0]
    g0[3] = (unsigned)((ga >> 32) & 0x1FFFFFFu) | (2u << 30);// global_addr[56:32] | type=2
    i32x8_t g1;
    g1[0] = 1 << 16;                                         // data_size=1 -> 2 bytes
    g1[1] = (int)((unsigned)(kElems & 0xFFFF) << 16);        // tensor_dim0[15:0]
    g1[2] = (int)(((unsigned)(kElems >> 16) & 0xFFFFu) |
                  ((unsigned)(rows & 0xFFFF) << 16));        // td0[31:16] | td1[15:0]
    g1[3] = (int)((unsigned)(kElems & 0xFFFF) << 16);        // td1[31:16]=0 | tile_dim0
    g1[4] = rows & 0xFFFF;                                   // tile_dim1 | tile_dim2=0
    g1[5] = kElems;                                          // tensor_dim0_stride[31:0]
    g1[6] = 0;                                               // stride hi | dim1_stride lo
    g1[7] = 0;
    const i32x4_t z4 = {0, 0, 0, 0};
    const i32x8_t z8 = {0, 0, 0, 0, 0, 0, 0, 0};
    __builtin_amdgcn_tensor_load_to_lds(g0, g1, z4, z4, z8, 0);
}

__device__ __forceinline__ unsigned lds_offset(const void* p) {
    return (unsigned)(unsigned long long)p;  // generic LDS addr: low 32 bits = offset
}
#else
#define FP_HAVE_TDM 0
#endif

// ---------------------------------------------------------------------------
// 0) f32 -> bf16 conversion (weights)
// ---------------------------------------------------------------------------
__global__ __launch_bounds__(256) void cvt_bf16_kernel(
    const float* __restrict__ src, unsigned short* __restrict__ dst, int n) {
    int i = blockIdx.x * 256 + threadIdx.x;
    if (i < n) dst[i] = f2bf(src[i]);
}

// ---------------------------------------------------------------------------
// 1) 3-NN inverse-distance interpolation + concat -> X bf16 [M x 384]
//    grid: B * (N2/256) blocks of 256 threads; one query point per thread
// ---------------------------------------------------------------------------
__global__ __launch_bounds__(256) void knn_interp_concat_kernel(
    const float* __restrict__ points1,   // [B,3,N1]
    const float* __restrict__ points2,   // [B,3,N2]
    const float* __restrict__ features1, // [B,C1,N1]
    const float* __restrict__ features2, // [B,C2,N2]
    unsigned short* __restrict__ xA)     // [M, CIN] bf16 row-major
{
    __shared__ float px[N1_], py[N1_], pz[N1_];
    const int b     = blockIdx.x >> 4;   // N2/256 = 16 chunks per batch
    const int chunk = blockIdx.x & 15;
    const int t     = threadIdx.x;

    const float* p1b = points1 + (size_t)b * 3 * N1_;
    for (int i = t; i < N1_; i += 256) {
        px[i] = p1b[i];
        py[i] = p1b[N1_ + i];
        pz[i] = p1b[2 * N1_ + i];
    }
    __syncthreads();

    const int n = chunk * 256 + t;
    const float* p2b = points2 + (size_t)b * 3 * N2_;
    const float qx = p2b[n];
    const float qy = p2b[N2_ + n];
    const float qz = p2b[2 * N2_ + n];

    float d0 = 3.4e38f, d1 = 3.4e38f, d2 = 3.4e38f;
    int   i0 = 0, i1 = 0, i2 = 0;
    for (int j = 0; j < N1_; ++j) {
        const float dx = qx - px[j], dy = qy - py[j], dz = qz - pz[j];
        const float d  = dx * dx + dy * dy + dz * dz;
        if (d < d2) {
            if (d < d1) {
                d2 = d1; i2 = i1;
                if (d < d0) { d1 = d0; i1 = i0; d0 = d; i0 = j; }
                else        { d1 = d;  i1 = j; }
            } else { d2 = d; i2 = j; }
        }
    }
    d0 = fmaxf(d0, 1e-10f); d1 = fmaxf(d1, 1e-10f); d2 = fmaxf(d2, 1e-10f);
    float w0 = 1.0f / d0, w1 = 1.0f / d1, w2 = 1.0f / d2;
    const float wn = 1.0f / (w0 + w1 + w2);
    w0 *= wn; w1 *= wn; w2 *= wn;

    const float* f1b = features1 + (size_t)b * C1_ * N1_;
    const float* f2b = features2 + (size_t)b * C2_ * N2_;
    unsigned short* xrow = xA + ((size_t)b * N2_ + n) * CIN_;

    #pragma unroll 4
    for (int c = 0; c < C1_; ++c) {
        const float* fc = f1b + (size_t)c * N1_;
        xrow[c] = f2bf(w0 * fc[i0] + w1 * fc[i1] + w2 * fc[i2]);
    }
    #pragma unroll 4
    for (int c = 0; c < C2_; ++c) {
        xrow[C1_ + c] = f2bf(f2b[(size_t)c * N2_ + n]);
    }
}

// ---------------------------------------------------------------------------
// 2) bf16 WMMA GEMM with LDS-staged panels (TDM when available):
//    Y[M,N] (f32) = A[M,K] (bf16 row-major) * W[N,K]^T        (N == 256)
//    Block (8 waves): 32(M) x 256(N) tile. Wave: 16 x 64, 4 accumulators.
//    LDS: sA [32 x K] + sB [256 x K] bf16, filled by tensor_load_to_lds.
//    grid: M/32 blocks of 256 threads
// ---------------------------------------------------------------------------
__global__ __launch_bounds__(256) void gemm_bf16_wmma_lds_kernel(
    const unsigned short* __restrict__ A,  // [M,K] bf16
    const unsigned short* __restrict__ W,  // [N,K] bf16 (output-channel major)
    float* __restrict__ Y,                 // [M,N] f32
    int M, int N, int K)
{
    extern __shared__ unsigned short smem[];
    unsigned short* sA = smem;                       // [32][K]
    unsigned short* sB = smem + 32 * (size_t)K;      // [N][K]

    const int m0 = blockIdx.x * 32;

#if FP_HAVE_TDM
    if (threadIdx.x < 32) {                 // wave 0 drives the DMA
        tdm_load_2d(A + (size_t)m0 * K, lds_offset(sA), K, 32);
        tdm_load_2d(W,                  lds_offset(sB), K, N);
        __builtin_amdgcn_s_wait_tensorcnt(0);
    }
    __syncthreads();
#else
    {
        const uint4* gB = (const uint4*)W;
        uint4*       lB = (uint4*)sB;
        const int nb = (N * K) >> 3;
        for (int i = threadIdx.x; i < nb; i += 256) lB[i] = gB[i];
        const uint4* gA = (const uint4*)(A + (size_t)m0 * K);
        uint4*       lA = (uint4*)sA;
        const int na = (32 * K) >> 3;
        for (int i = threadIdx.x; i < na; i += 256) lA[i] = gA[i];
    }
    __syncthreads();
#endif

    const int lane   = threadIdx.x & 31;
    const int wave   = threadIdx.x >> 5;
    const int nGroup = wave & 3;           // 4 column groups of 64
    const int mLoc   = (wave >> 2) * 16;   // 0 or 16 within the 32-row tile

    const int l15  = lane & 15;
    const int half = lane >> 4;            // 0: lanes 0-15, 1: lanes 16-31

    // A frag: contiguous 16 K/half;  B frag: contiguous 16 K/half
    const unsigned short* arow = sA + (size_t)(mLoc + l15) * K + half * 16;
    const int ncol = nGroup * 64 + l15;
    const unsigned short* b0 = sB + (size_t)(ncol +  0) * K + half * 16;
    const unsigned short* b1 = sB + (size_t)(ncol + 16) * K + half * 16;
    const unsigned short* b2 = sB + (size_t)(ncol + 32) * K + half * 16;
    const unsigned short* b3 = sB + (size_t)(ncol + 48) * K + half * 16;

    f32x8 acc0 = {}, acc1 = {}, acc2 = {}, acc3 = {};

    for (int k = 0; k < K; k += 32) {
        const bf16x16 af = ld_frag(arow + k, arow + k + 8);
        const bf16x16 f0 = ld_frag(b0 + k, b0 + k + 8);
        const bf16x16 f1 = ld_frag(b1 + k, b1 + k + 8);
        const bf16x16 f2 = ld_frag(b2 + k, b2 + k + 8);
        const bf16x16 f3 = ld_frag(b3 + k, b3 + k + 8);
        acc0 = __builtin_amdgcn_wmma_f32_16x16x32_bf16(false, af, false, f0,
                                                       (short)0, acc0, false, false);
        acc1 = __builtin_amdgcn_wmma_f32_16x16x32_bf16(false, af, false, f1,
                                                       (short)0, acc1, false, false);
        acc2 = __builtin_amdgcn_wmma_f32_16x16x32_bf16(false, af, false, f2,
                                                       (short)0, acc2, false, false);
        acc3 = __builtin_amdgcn_wmma_f32_16x16x32_bf16(false, af, false, f3,
                                                       (short)0, acc3, false, false);
    }

    // C/D layout: lane l15 = column, VGPR v = row (v + 8*half) within 16x16 tile
    const int colb = nGroup * 64 + l15;
    const int rowb = m0 + mLoc + half * 8;
    float* yb = Y + (size_t)rowb * N + colb;
    #pragma unroll
    for (int v = 0; v < 8; ++v) {
        yb[(size_t)v * N +  0] = acc0[v];
        yb[(size_t)v * N + 16] = acc1[v];
        yb[(size_t)v * N + 32] = acc2[v];
        yb[(size_t)v * N + 48] = acc3[v];
    }
}

// ---------------------------------------------------------------------------
// 3) BN stats, stage 1: per-block partial sums over a row chunk
// ---------------------------------------------------------------------------
__global__ __launch_bounds__(256) void bn_partial_kernel(
    const float* __restrict__ Y, float* __restrict__ psum,
    float* __restrict__ psumsq)
{
    const int c = threadIdx.x;
    const int rows = M_ / RED_BLOCKS;
    const size_t r0 = (size_t)blockIdx.x * rows;
    float s = 0.0f, s2 = 0.0f;
    for (int r = 0; r < rows; ++r) {
        const float v = Y[(r0 + r) * H_ + c];
        s  += v;
        s2 += v * v;
    }
    psum  [blockIdx.x * H_ + c] = s;
    psumsq[blockIdx.x * H_ + c] = s2;
}

// ---------------------------------------------------------------------------
// 4) BN stats, stage 2: finalize -> per-channel affine (scale, shift)
//    (conv bias cancels exactly under training-mode BN, so it is omitted)
// ---------------------------------------------------------------------------
__global__ __launch_bounds__(256) void bn_finalize_kernel(
    const float* __restrict__ psum, const float* __restrict__ psumsq,
    const float* __restrict__ g, const float* __restrict__ beta,
    float* __restrict__ scale, float* __restrict__ shift)
{
    const int c = threadIdx.x;
    float s = 0.0f, s2 = 0.0f;
    for (int i = 0; i < RED_BLOCKS; ++i) {
        s  += psum  [i * H_ + c];
        s2 += psumsq[i * H_ + c];
    }
    const float invM = 1.0f / (float)M_;
    const float mean = s * invM;
    const float var  = s2 * invM - mean * mean;
    const float a    = g[c] * rsqrtf(var + 1e-3f);
    scale[c] = a;
    shift[c] = beta[c] - mean * a;
}

// ---------------------------------------------------------------------------
// 5a) normalize + ReLU -> bf16 activations for the next GEMM
// ---------------------------------------------------------------------------
__global__ __launch_bounds__(256) void bn_apply_bf16_kernel(
    const float* __restrict__ Y, const float* __restrict__ scale,
    const float* __restrict__ shift, unsigned short* __restrict__ X2)
{
    const int c = threadIdx.x;
    const size_t m = blockIdx.x;
    const float v = fmaxf(Y[m * H_ + c] * scale[c] + shift[c], 0.0f);
    X2[m * H_ + c] = f2bf(v);
}

// ---------------------------------------------------------------------------
// 5b) normalize + ReLU -> final output f32 [B, H, N2]
// ---------------------------------------------------------------------------
__global__ __launch_bounds__(256) void bn_apply_out_kernel(
    const float* __restrict__ Y, const float* __restrict__ scale,
    const float* __restrict__ shift, float* __restrict__ out)
{
    const int c = threadIdx.x;
    const size_t m = blockIdx.x;
    const int b = (int)(m >> 12);        // N2 = 4096
    const int n = (int)(m & 4095);
    const float v = fmaxf(Y[m * H_ + c] * scale[c] + shift[c], 0.0f);
    out[((size_t)b * H_ + c) * N2_ + n] = v;
}

// ---------------------------------------------------------------------------
// host-side orchestration
// ---------------------------------------------------------------------------
extern "C" void kernel_launch(void* const* d_in, const int* in_sizes, int n_in,
                              void* d_out, int out_size, void* d_ws, size_t ws_size,
                              hipStream_t stream) {
    const float* points1   = (const float*)d_in[0];
    const float* points2   = (const float*)d_in[1];
    const float* features1 = (const float*)d_in[2];
    const float* features2 = (const float*)d_in[3];
    const float* W1        = (const float*)d_in[4];
    // d_in[5] = b1 (cancels under BN)
    const float* g1        = (const float*)d_in[6];
    const float* beta1     = (const float*)d_in[7];
    const float* W2        = (const float*)d_in[8];
    // d_in[9] = b2 (cancels under BN)
    const float* g2        = (const float*)d_in[10];
    const float* beta2     = (const float*)d_in[11];
    float* out = (float*)d_out;

    auto align256 = [](size_t x) { return (x + 255) & ~(size_t)255; };
    char* w = (char*)d_ws;
    size_t off = 0;
    unsigned short* xA   = (unsigned short*)(w + off); off += align256((size_t)M_ * CIN_ * 2);
    float*          Y    = (float*)         (w + off); off += align256((size_t)M_ * H_ * 4);
    unsigned short* X2   = (unsigned short*)(w + off); off += align256((size_t)M_ * H_ * 2);
    unsigned short* W1b  = (unsigned short*)(w + off); off += align256((size_t)H_ * CIN_ * 2);
    unsigned short* W2b  = (unsigned short*)(w + off); off += align256((size_t)H_ * H_ * 2);
    float*          psum = (float*)         (w + off); off += align256((size_t)RED_BLOCKS * H_ * 4);
    float*          psq  = (float*)         (w + off); off += align256((size_t)RED_BLOCKS * H_ * 4);
    float*          scl  = (float*)         (w + off); off += align256((size_t)H_ * 4);
    float*          shf  = (float*)         (w + off); off += align256((size_t)H_ * 4);
    (void)ws_size; (void)n_in; (void)in_sizes; (void)out_size;

    // weights -> bf16 (already in [O, K] = B-fragment-friendly layout)
    cvt_bf16_kernel<<<(H_ * CIN_ + 255) / 256, 256, 0, stream>>>(W1, W1b, H_ * CIN_);
    cvt_bf16_kernel<<<(H_ * H_   + 255) / 256, 256, 0, stream>>>(W2, W2b, H_ * H_);

    // 3-NN interpolation + concat -> X [M x 384] bf16
    knn_interp_concat_kernel<<<B_ * (N2_ / 256), 256, 0, stream>>>(
        points1, points2, features1, features2, xA);

    // layer 1: GEMM (K=384) + BN + ReLU -> bf16
    {
        const size_t lds = (size_t)(32 + H_) * CIN_ * 2;   // 216 KB (of 320 KB/WGP)
        gemm_bf16_wmma_lds_kernel<<<M_ / 32, 256, lds, stream>>>(xA, W1b, Y, M_, H_, CIN_);
        bn_partial_kernel<<<RED_BLOCKS, 256, 0, stream>>>(Y, psum, psq);
        bn_finalize_kernel<<<1, 256, 0, stream>>>(psum, psq, g1, beta1, scl, shf);
        bn_apply_bf16_kernel<<<M_, 256, 0, stream>>>(Y, scl, shf, X2);
    }

    // layer 2: GEMM (K=256) + BN + ReLU -> output [B, 256, N2] f32
    {
        const size_t lds = (size_t)(32 + H_) * H_ * 2;     // 144 KB
        gemm_bf16_wmma_lds_kernel<<<M_ / 32, 256, lds, stream>>>(X2, W2b, Y, M_, H_, H_);
        bn_partial_kernel<<<RED_BLOCKS, 256, 0, stream>>>(Y, psum, psq);
        bn_finalize_kernel<<<1, 256, 0, stream>>>(psum, psq, g2, beta2, scl, shf);
        bn_apply_out_kernel<<<M_, 256, 0, stream>>>(Y, scl, shf, out);
    }
}
